// DepParser_88983132438737
// MI455X (gfx1250) — matrix operand order repaired and checked by reading, not compiled, not run on hardware
//
#include <hip/hip_runtime.h>
#include <cstdint>

// ---------------------------------------------------------------------------
// Dependency-parser forward pass for gfx1250 (MI455X).
//   phase 1: bf16 conversions / packing of weights (one-time, tiny)
//   phase 2: WMMA bf16 GEMMs for LSTM input gates
//   phase 3: serial LSTM recurrence (2 workgroups, h in LDS, Whh^T in L2)
//   phase 4: WMMA GEMMs for arc projections s1/s2 and label MLP
//   phase 5: 257x257x512 tanh-reduction arc scorer (LDS-tiled, conflict-free)
// ---------------------------------------------------------------------------

#define N_SENT 256
#define HID    512     // H
#define GATES  2048    // 4H
#define TWOH   1024    // 2H
#define MROWS  257     // n+1
#define MPAD   272     // 17 * 16
#define LBL    40
#define LBLPAD 48

typedef __bf16 bf16_t;
typedef bf16_t v16bf __attribute__((ext_vector_type(16)));
typedef bf16_t v8bf  __attribute__((ext_vector_type(8)));
typedef float  v8f   __attribute__((ext_vector_type(8)));

__device__ __forceinline__ uint16_t f2bf(float f) {
  uint32_t u = __float_as_uint(f);
  u += 0x7fffu + ((u >> 16) & 1u);   // round-to-nearest-even
  return (uint16_t)(u >> 16);
}
__device__ __forceinline__ float bflo(uint32_t u) { return __uint_as_float(u << 16); }
__device__ __forceinline__ float bfhi(uint32_t u) { return __uint_as_float(u & 0xffff0000u); }
__device__ __forceinline__ float fsigmoid(float x) { return 1.0f / (1.0f + __expf(-x)); }
__device__ __forceinline__ float ftanh(float x) {
  x = fminf(fmaxf(x, -15.0f), 15.0f);
  float e = __expf(2.0f * x);
  return (e - 1.0f) / (e + 1.0f);
}

// ------------------------- small helper kernels ----------------------------

__global__ void cvt_bf16_kernel(const float* __restrict__ src,
                                uint16_t* __restrict__ dst, int n) {
  for (int i = blockIdx.x * blockDim.x + threadIdx.x; i < n;
       i += gridDim.x * blockDim.x)
    dst[i] = f2bf(src[i]);
}

__global__ void zero_f32_kernel(float* __restrict__ p, int n) {
  int i = blockIdx.x * blockDim.x + threadIdx.x;
  if (i < n) p[i] = 0.0f;
}

__global__ void bias_sum_kernel(const float* __restrict__ a,
                                const float* __restrict__ b,
                                float* __restrict__ o, int n) {
  int i = blockIdx.x * blockDim.x + threadIdx.x;
  if (i < n) o[i] = a[i] + b[i];
}

// Whh [2048,512] fp32 -> WhhT2 uint32 [256,2048]; dword j at row k2 packs
// bf16(Whh[j, 2k2]) in [15:0] and bf16(Whh[j, 2k2+1]) in [31:16].
__global__ void pack_whh_kernel(const float* __restrict__ whh,
                                uint32_t* __restrict__ dst) {
  int k2 = blockIdx.x;  // 0..255
  for (int j = threadIdx.x; j < GATES; j += blockDim.x) {
    uint16_t lo = f2bf(whh[(size_t)j * HID + 2 * k2]);
    uint16_t hi = f2bf(whh[(size_t)j * HID + 2 * k2 + 1]);
    dst[(size_t)k2 * GATES + j] = (uint32_t)lo | ((uint32_t)hi << 16);
  }
}

// mlp2_w [40,512] -> bf16 [48,512], zero-padded rows 40..47
__global__ void cvt_mlp2_kernel(const float* __restrict__ src,
                                uint16_t* __restrict__ dst) {
  int i = blockIdx.x * blockDim.x + threadIdx.x;  // 0 .. 48*512-1
  if (i < LBLPAD * HID) {
    int r = i / HID;
    dst[i] = (r < LBL) ? f2bf(src[i]) : (uint16_t)0;
  }
}

// x[t] = [W_emb[words[t]], P_emb[pos[t]]]  (bf16), plus reversed copy
__global__ void embed_kernel(const int* __restrict__ words,
                             const int* __restrict__ pos,
                             const float* __restrict__ W_emb,
                             const float* __restrict__ P_emb,
                             uint16_t* __restrict__ x_bf,
                             uint16_t* __restrict__ xr_bf) {
  int t = blockIdx.x;      // 0..255
  int c = threadIdx.x;     // 0..511
  float v = (c < 256) ? W_emb[(size_t)words[t] * 256 + c]
                      : P_emb[(size_t)pos[t] * 256 + (c - 256)];
  uint16_t bv = f2bf(v);
  x_bf[(size_t)t * HID + c] = bv;
  xr_bf[(size_t)(N_SENT - 1 - t) * HID + c] = bv;
}

// Ls row i = [out[gl[i,0]], out[gl[i,1]]]  (bf16)
__global__ void gather_ls_kernel(const int* __restrict__ gl,
                                 const float* __restrict__ out,
                                 uint16_t* __restrict__ ls_bf) {
  int i = blockIdx.x;  // 0..255
  for (int c = threadIdx.x; c < 2 * TWOH; c += blockDim.x) {
    int which = c >> 10;                       // 0: head, 1: dep
    int row = gl[i * 3 + which];               // in [0, 257)
    float v = out[(size_t)row * TWOH + (c & (TWOH - 1))];
    ls_bf[(size_t)i * (2 * TWOH) + c] = f2bf(v);
  }
}

// ------------------------------ WMMA GEMM ----------------------------------
// C[M,N] = act( sum_k A[m,k] * B[n,k] + bias[n] )   (A,B bf16, fp32 acc)
// Block = 256 thr = 8 waves; wave w owns one 16x16 N-tile, block row = M-tile.
// Fragment layouts per CDNA5 ISA 7.12.2 (16-bit A 16x32, B 32x16, f32 C/D).
template <int TANH, int BF16OUT>
__global__ __launch_bounds__(256) void wmma_gemm_kernel(
    const uint16_t* __restrict__ A, const uint16_t* __restrict__ B,
    const float* __restrict__ bias, void* __restrict__ Cp, int lda, int ldb,
    int ldc, int K, int Npad, int Mstore, int Nstore) {
  const int lane = threadIdx.x & 31;
  const int wave = threadIdx.x >> 5;
  const int ntile = blockIdx.x * 8 + wave;
  if (ntile * 16 >= Npad) return;  // wave-uniform: EXEC stays all-ones
  const int m0 = blockIdx.y * 16;
  const int n0 = ntile * 16;
  const int idx  = lane & 15;   // A: row M, B: col N, C: col N
  const int half = lane >> 4;   // K-half selector / C row-half

  float bn = 0.0f;
  if (bias && (n0 + idx) < Nstore) bn = bias[n0 + idx];
  v8f acc;
#pragma unroll
  for (int r = 0; r < 8; ++r) acc[r] = bn;

  const uint16_t* Arow = A + (size_t)(m0 + idx) * lda;
  const uint16_t* Brow = B + (size_t)(n0 + idx) * ldb;
  for (int k0 = 0; k0 < K; k0 += 32) {
    // A lane: K = half*8 + {0..7}  and  16 + half*8 + {0..7}
    v8bf a0 = *(const v8bf*)(Arow + k0 + half * 8);
    v8bf a1 = *(const v8bf*)(Arow + k0 + 16 + half * 8);
    // B lane: K = half*8 + {0..7}  and  16 + half*8 + {0..7}
    v8bf b0 = *(const v8bf*)(Brow + k0 + half * 8);
    v8bf b1 = *(const v8bf*)(Brow + k0 + 16 + half * 8);
    v16bf av = __builtin_shufflevector(a0, a1, 0, 1, 2, 3, 4, 5, 6, 7, 8, 9,
                                       10, 11, 12, 13, 14, 15);
    v16bf bv = __builtin_shufflevector(b0, b1, 0, 1, 2, 3, 4, 5, 6, 7, 8, 9,
                                       10, 11, 12, 13, 14, 15);
    acc = __builtin_amdgcn_wmma_f32_16x16x32_bf16(false, av, false, bv,
                                                  (short)0, acc, false, false);
  }

  const int col = n0 + idx;
  if (col < Nstore) {
#pragma unroll
    for (int r = 0; r < 8; ++r) {
      int m = m0 + r + half * 8;  // C/D: VGPR r -> M=r (lanes<16) / 8+r
      if (m < Mstore) {
        float v = acc[r];
        if (TANH) v = ftanh(v);
        if (BF16OUT)
          ((uint16_t*)Cp)[(size_t)m * ldc + col] = f2bf(v);
        else
          ((float*)Cp)[(size_t)m * ldc + col] = v;
      }
    }
  }
}

// ----------------------------- LSTM recurrence -----------------------------
// grid = 2 (fwd/bwd), block = 512 (one thread per hidden unit).
// Gates j = {u, H+u, 2H+u, 3H+u}; Whh^T packed 2 bf16 per dword, coalesced.
__global__ __launch_bounds__(512) void lstm_kernel(
    const float* __restrict__ xg_f, const float* __restrict__ xg_b,
    const uint32_t* __restrict__ wTf, const uint32_t* __restrict__ wTb,
    float* __restrict__ out) {
  const int dir = blockIdx.x;
  const float* xg = dir ? xg_b : xg_f;
  const uint32_t* wT = dir ? wTb : wTf;
  const int u = threadIdx.x;
  __shared__ float sh[HID];
  sh[u] = 0.0f;
  float c = 0.0f;
  __syncthreads();
  for (int t = 0; t < N_SENT; ++t) {
    const float* g = xg + (size_t)t * GATES;
    float ai = g[u], af = g[HID + u], ag = g[2 * HID + u], ao = g[3 * HID + u];
#pragma unroll 4
    for (int k2 = 0; k2 < HID / 2; ++k2) {
      float h0 = sh[2 * k2];          // LDS broadcast
      float h1 = sh[2 * k2 + 1];
      const uint32_t* row = wT + (size_t)k2 * GATES + u;  // coalesced
      uint32_t wi = row[0], wf = row[HID], wg = row[2 * HID], wo = row[3 * HID];
      ai += h0 * bflo(wi) + h1 * bfhi(wi);
      af += h0 * bflo(wf) + h1 * bfhi(wf);
      ag += h0 * bflo(wg) + h1 * bfhi(wg);
      ao += h0 * bflo(wo) + h1 * bfhi(wo);
    }
    c = fsigmoid(af) * c + fsigmoid(ai) * ftanh(ag);
    float h = fsigmoid(ao) * ftanh(c);
    __syncthreads();
    sh[u] = h;
    __syncthreads();
    if (dir == 0)
      out[(size_t)(t + 1) * TWOH + u] = h;             // hf[t]
    else
      out[(size_t)(N_SENT - t) * TWOH + HID + u] = h;  // hb[255-t] un-reversed
  }
}

// ------------------------------ arc scorer ---------------------------------
// M[i,j] = sum_h tanh(s1[i,h] + s2[j,h]) * w[h] + b.  16x16 (i,j) tile/block.
// LDS row stride 260 (260 % 64 == 4) -> 16 distinct banks across j lanes.
__global__ __launch_bounds__(256) void arc_pair_kernel(
    const float* __restrict__ s1, const float* __restrict__ s2,
    const float* __restrict__ w, const float* __restrict__ b,
    float* __restrict__ Mout) {
  __shared__ float l1[16 * 260];
  __shared__ float l2[16 * 260];
  __shared__ float wl[256];
  const int tid = threadIdx.x;
  const int i0 = blockIdx.y * 16, j0 = blockIdx.x * 16;
  const int il = tid >> 4, jl = tid & 15;
  float acc = 0.0f;
  for (int ph = 0; ph < 2; ++ph) {
    __syncthreads();
#pragma unroll
    for (int e = 0; e < 16; ++e) {  // row e, col tid: coalesced + no conflicts
      l1[e * 260 + tid] = s1[(size_t)(i0 + e) * HID + ph * 256 + tid];
      l2[e * 260 + tid] = s2[(size_t)(j0 + e) * HID + ph * 256 + tid];
    }
    wl[tid] = w[ph * 256 + tid];
    __syncthreads();
    for (int h = 0; h < 256; ++h)
      acc += ftanh(l1[il * 260 + h] + l2[jl * 260 + h]) * wl[h];
  }
  const int i = i0 + il, j = j0 + jl;
  if (i < MROWS && j < MROWS) Mout[(size_t)i * MROWS + j] = acc + b[0];
}

// ------------------------------- launcher ----------------------------------

extern "C" void kernel_launch(void* const* d_in, const int* in_sizes, int n_in,
                              void* d_out, int out_size, void* d_ws,
                              size_t ws_size, hipStream_t stream) {
  (void)in_sizes; (void)n_in; (void)out_size; (void)ws_size;
  const int*   words  = (const int*)d_in[0];
  const int*   pos    = (const int*)d_in[1];
  const int*   gl     = (const int*)d_in[2];
  const float* W_emb  = (const float*)d_in[3];
  const float* P_emb  = (const float*)d_in[4];
  const float* Wih_f  = (const float*)d_in[5];
  const float* Whh_f  = (const float*)d_in[6];
  const float* bih_f  = (const float*)d_in[7];
  const float* bhh_f  = (const float*)d_in[8];
  const float* Wih_b  = (const float*)d_in[9];
  const float* Whh_b  = (const float*)d_in[10];
  const float* bih_b  = (const float*)d_in[11];
  const float* bhh_b  = (const float*)d_in[12];
  const float* fc1_w  = (const float*)d_in[13];
  const float* fc1_b  = (const float*)d_in[14];
  const float* fc2_w  = (const float*)d_in[15];
  const float* fc2_b  = (const float*)d_in[16];
  const float* mlp1_w = (const float*)d_in[17];
  const float* mlp1_b = (const float*)d_in[18];
  const float* mlp2_w = (const float*)d_in[19];
  const float* mlp2_b = (const float*)d_in[20];

  char* ws = (char*)d_ws;
  size_t off = 0;
  auto alloc = [&](size_t bytes) {
    size_t o = off;
    off += (bytes + 255) & ~(size_t)255;
    return o;
  };
  uint16_t* x_bf     = (uint16_t*)(ws + alloc((size_t)N_SENT * HID * 2));
  uint16_t* xr_bf    = (uint16_t*)(ws + alloc((size_t)N_SENT * HID * 2));
  uint16_t* wihf_bf  = (uint16_t*)(ws + alloc((size_t)GATES * HID * 2));
  uint16_t* wihb_bf  = (uint16_t*)(ws + alloc((size_t)GATES * HID * 2));
  uint32_t* whhT2_f  = (uint32_t*)(ws + alloc((size_t)(HID / 2) * GATES * 4));
  uint32_t* whhT2_b  = (uint32_t*)(ws + alloc((size_t)(HID / 2) * GATES * 4));
  float*    bias_f   = (float*)(ws + alloc((size_t)GATES * 4));
  float*    bias_b   = (float*)(ws + alloc((size_t)GATES * 4));
  float*    xg_f     = (float*)(ws + alloc((size_t)N_SENT * GATES * 4));
  float*    xg_b     = (float*)(ws + alloc((size_t)N_SENT * GATES * 4));
  float*    out_f32  = (float*)(ws + alloc((size_t)MPAD * TWOH * 4));
  uint16_t* out_bf   = (uint16_t*)(ws + alloc((size_t)MPAD * TWOH * 2));
  uint16_t* fc1_bf   = (uint16_t*)(ws + alloc((size_t)HID * GATES * 2));
  float*    s1       = (float*)(ws + alloc((size_t)MPAD * HID * 4));
  float*    s2       = (float*)(ws + alloc((size_t)MPAD * HID * 4));
  uint16_t* ls_bf    = (uint16_t*)(ws + alloc((size_t)N_SENT * GATES * 2));
  uint16_t* mlp1_bf  = (uint16_t*)(ws + alloc((size_t)HID * GATES * 2));
  uint16_t* tlab_bf  = (uint16_t*)(ws + alloc((size_t)N_SENT * HID * 2));
  uint16_t* mlp2_bf  = (uint16_t*)(ws + alloc((size_t)LBLPAD * HID * 2));

  float* Mout = (float*)d_out;                 // [257*257]
  float* Lout = (float*)d_out + MROWS * MROWS; // [256*40]

  // ---- phase 1: weight conversions ----
  cvt_bf16_kernel<<<1024, 256, 0, stream>>>(Wih_f, wihf_bf, GATES * HID);
  cvt_bf16_kernel<<<1024, 256, 0, stream>>>(Wih_b, wihb_bf, GATES * HID);
  cvt_bf16_kernel<<<1024, 256, 0, stream>>>(fc1_w, fc1_bf, HID * GATES);
  cvt_bf16_kernel<<<1024, 256, 0, stream>>>(mlp1_w, mlp1_bf, HID * GATES);
  pack_whh_kernel<<<HID / 2, 256, 0, stream>>>(Whh_f, whhT2_f);
  pack_whh_kernel<<<HID / 2, 256, 0, stream>>>(Whh_b, whhT2_b);
  bias_sum_kernel<<<GATES / 256, 256, 0, stream>>>(bih_f, bhh_f, bias_f, GATES);
  bias_sum_kernel<<<GATES / 256, 256, 0, stream>>>(bih_b, bhh_b, bias_b, GATES);
  cvt_mlp2_kernel<<<(LBLPAD * HID + 255) / 256, 256, 0, stream>>>(mlp2_w, mlp2_bf);
  embed_kernel<<<N_SENT, HID, 0, stream>>>(words, pos, W_emb, P_emb, x_bf, xr_bf);
  zero_f32_kernel<<<(MPAD * TWOH + 255) / 256, 256, 0, stream>>>(out_f32, MPAD * TWOH);

  // ---- phase 2: input-gate GEMMs  xg = x @ Wih^T + (bih+bhh) ----
  wmma_gemm_kernel<0, 0><<<dim3(GATES / 128, N_SENT / 16), 256, 0, stream>>>(
      x_bf, wihf_bf, bias_f, xg_f, HID, HID, GATES, HID, GATES, N_SENT, GATES);
  wmma_gemm_kernel<0, 0><<<dim3(GATES / 128, N_SENT / 16), 256, 0, stream>>>(
      xr_bf, wihb_bf, bias_b, xg_b, HID, HID, GATES, HID, GATES, N_SENT, GATES);

  // ---- phase 3: recurrence (both directions concurrently) ----
  lstm_kernel<<<2, HID, 0, stream>>>(xg_f, xg_b, whhT2_f, whhT2_b, out_f32);
  cvt_bf16_kernel<<<(MPAD * TWOH + 255) / 256, 256, 0, stream>>>(
      out_f32, out_bf, MPAD * TWOH);

  // ---- phase 4: arc projections  s1 = out@A1 + fc1_b, s2 = out@A2 ----
  wmma_gemm_kernel<0, 0><<<dim3(HID / 128, MPAD / 16), 256, 0, stream>>>(
      out_bf, fc1_bf, fc1_b, s1, TWOH, GATES, HID, TWOH, HID, MPAD, HID);
  wmma_gemm_kernel<0, 0><<<dim3(HID / 128, MPAD / 16), 256, 0, stream>>>(
      out_bf, fc1_bf + TWOH, nullptr, s2, TWOH, GATES, HID, TWOH, HID, MPAD, HID);

  // ---- phase 5: all-pairs arc scores ----
  arc_pair_kernel<<<dim3(MPAD / 16, MPAD / 16), 256, 0, stream>>>(
      s1, s2, fc2_w, fc2_b, Mout);

  // ---- label path: T = tanh(Ls@mlp1^T + b1);  L = T@mlp2^T + b2 ----
  gather_ls_kernel<<<N_SENT, 256, 0, stream>>>(gl, out_f32, ls_bf);
  wmma_gemm_kernel<1, 1><<<dim3(HID / 128, N_SENT / 16), 256, 0, stream>>>(
      ls_bf, mlp1_bf, mlp1_b, tlab_bf, GATES, GATES, HID, GATES, HID, N_SENT, HID);
  wmma_gemm_kernel<0, 0><<<dim3(1, N_SENT / 16), 256, 0, stream>>>(
      tlab_bf, mlp2_bf, mlp2_b, Lout, HID, HID, LBL, HID, LBLPAD, N_SENT, LBL);
}
